// TLencoder_62809601736757
// MI455X (gfx1250) — matrix operand order repaired and checked by reading, not compiled, not run on hardware
//
#include <hip/hip_runtime.h>
#include <hip/hip_bf16.h>
#include <math.h>

typedef __bf16 bf16_t;
typedef __bf16 v16bf __attribute__((ext_vector_type(16)));
typedef __bf16 v8bf  __attribute__((ext_vector_type(8)));
typedef float  v8f   __attribute__((ext_vector_type(8)));
typedef int    v4i   __attribute__((ext_vector_type(4)));

typedef __attribute__((address_space(1))) v4i* gptr_v4i;
typedef __attribute__((address_space(3))) v4i* lptr_v4i;

#define BSZ 16
#define SEQ 256
#define DIM 512
#define NH  8
#define DHD 64
#define DFF 2048
#define TOK (BSZ*SEQ)

enum { MODE_F32 = 0, MODE_BF16 = 1, MODE_GELU = 2, MODE_F32BIAS = 3, MODE_KROW = 4, MODE_VT = 5 };

#if __has_builtin(__builtin_amdgcn_global_load_async_to_lds_b128)
#define USE_ASYNC_LDS 1
#endif

__device__ __forceinline__ void wait_asynccnt0() {
#if __has_builtin(__builtin_amdgcn_s_wait_asynccnt)
  __builtin_amdgcn_s_wait_asynccnt(0);
#else
  asm volatile("s_wait_asynccnt 0x0" ::: "memory");
#endif
}

__device__ __forceinline__ float gelu_exact(float x) {
  return 0.5f * x * (1.0f + erff(x * 0.70710678118654752f));
}

// pack two contiguous 8-elem runs into a WMMA A/B fragment register set
__device__ __forceinline__ v16bf pack_frag(v8bf lo, v8bf hi) {
  v16bf r;
#pragma unroll
  for (int e = 0; e < 8; ++e) { r[e] = lo[e]; r[8 + e] = hi[e]; }
  return r;
}

// ---------------- elementwise converts ----------------
__global__ void convert_f32_bf16(const float* __restrict__ in, bf16_t* __restrict__ out, int n) {
  int i = blockIdx.x * blockDim.x + threadIdx.x;
  if (i < n) out[i] = (bf16_t)in[i];
}

// feature-dim conv, kernel 3, zero pad, f32 in -> bf16 out
__global__ void conv3_feature(const float* __restrict__ X, const float* __restrict__ w,
                              bf16_t* __restrict__ Ob, int rows) {
  int i = blockIdx.x * blockDim.x + threadIdx.x;
  if (i >= rows * DIM) return;
  int t = i >> 9, d = i & (DIM - 1);
  const float* xr = X + (size_t)t * DIM;
  float v = w[1] * xr[d];
  if (d > 0)       v += w[0] * xr[d - 1];
  if (d < DIM - 1) v += w[2] * xr[d + 1];
  Ob[i] = (bf16_t)v;
}

// ---------------- blocked WMMA GEMM: C[MxN] = A[MxK] @ B[KxN] (+epilogue) ----------------
// block tile 128x64, 8 waves, each wave owns 32x32 (2x2 wmma tiles). K step 32.
__global__ __launch_bounds__(256)
void gemm_wmma(const bf16_t* __restrict__ A, const bf16_t* __restrict__ B,
               const float* __restrict__ bias, void* __restrict__ Cout,
               int M, int N, int K, int mode) {
  __shared__ alignas(16) bf16_t sA[128][32];   // row-major panel, 8KB
  __shared__ alignas(16) bf16_t sBT[64][32];   // transposed (n-major) panel, 4KB
  const int tid  = threadIdx.x;
  const int lane = tid & 31;
  const int wave = tid >> 5;
  const int nBlkN = N >> 6;
  const int m0 = (int)(blockIdx.x / nBlkN) << 7;
  const int n0 = (int)(blockIdx.x % nBlkN) << 6;
  const int mw = (wave >> 1) << 5;
  const int nw = (wave & 1) << 5;
  const int mrow  = lane & 15;
  const int khalf = lane >> 4;

  // staging maps
  const int arow = tid >> 1;           // 0..127
  const int acol = (tid & 1) << 4;     // 0 / 16
  const int bk   = tid >> 3;           // 0..31
  const int bn   = (tid & 7) << 3;     // 0..56

  v8f acc[2][2] = {};

  for (int k0 = 0; k0 < K; k0 += 32) {
    __syncthreads();
    // ---- stage A panel: global -> LDS (async DMA path if available) ----
    const bf16_t* gA = A + (size_t)(m0 + arow) * K + k0 + acol;
    bf16_t* lA = &sA[arow][acol];
#ifdef USE_ASYNC_LDS
    __builtin_amdgcn_global_load_async_to_lds_b128(
        (gptr_v4i)(void*)gA, (lptr_v4i)(void*)lA, 0, 0);
    __builtin_amdgcn_global_load_async_to_lds_b128(
        (gptr_v4i)(void*)gA, (lptr_v4i)(void*)lA, 16, 0);
#else
    *(v8bf*)lA       = *(const v8bf*)gA;
    *(v8bf*)(lA + 8) = *(const v8bf*)(gA + 8);
#endif
    // ---- stage B panel transposed via VGPRs ----
    {
      v8bf bvec = *(const v8bf*)(B + (size_t)(k0 + bk) * N + n0 + bn);
#pragma unroll
      for (int e = 0; e < 8; ++e) sBT[bn + e][bk] = bvec[e];
    }
    if (k0 + 32 < K) {  // prefetch next panels into cache
      __builtin_prefetch(A + (size_t)(m0 + arow) * K + k0 + 32 + acol, 0, 1);
      __builtin_prefetch(B + (size_t)(k0 + 32 + bk) * N + n0 + bn, 0, 1);
    }
#ifdef USE_ASYNC_LDS
    wait_asynccnt0();
#endif
    __syncthreads();

    // ---- fragments from LDS (all 16B ds loads) + 2x2 WMMA ----
    v16bf af[2], bfr[2];
#pragma unroll
    for (int i = 0; i < 2; ++i) {
      const bf16_t* ar = &sA[mw + (i << 4) + mrow][0];
      af[i] = pack_frag(*(const v8bf*)(ar + (khalf << 3)),
                        *(const v8bf*)(ar + 16 + (khalf << 3)));
    }
#pragma unroll
    for (int j = 0; j < 2; ++j) {
      const bf16_t* br = &sBT[nw + (j << 4) + mrow][khalf << 3];
      bfr[j] = pack_frag(*(const v8bf*)br, *(const v8bf*)(br + 16));
    }
#pragma unroll
    for (int i = 0; i < 2; ++i)
#pragma unroll
      for (int j = 0; j < 2; ++j)
        acc[i][j] = __builtin_amdgcn_wmma_f32_16x16x32_bf16(false, af[i], false, bfr[j],
                                                            (short)0, acc[i][j], false, false);
  }

  // ---- epilogue ----
#pragma unroll
  for (int i = 0; i < 2; ++i) {
#pragma unroll
    for (int j = 0; j < 2; ++j) {
      const int col = n0 + nw + (j << 4) + mrow;
      const float bval = (mode == MODE_GELU || mode == MODE_F32BIAS) ? bias[col] : 0.f;
#pragma unroll
      for (int r = 0; r < 8; ++r) {
        const int row = m0 + mw + (i << 4) + r + (khalf << 3);
        const float vv = acc[i][j][r];
        if (mode == MODE_F32) {
          ((float*)Cout)[(size_t)row * N + col] = vv;
        } else if (mode == MODE_BF16) {
          ((bf16_t*)Cout)[(size_t)row * N + col] = (bf16_t)vv;
        } else if (mode == MODE_GELU) {
          ((bf16_t*)Cout)[(size_t)row * N + col] = (bf16_t)gelu_exact(vv + bval);
        } else if (mode == MODE_F32BIAS) {
          ((float*)Cout)[(size_t)row * N + col] = vv + bval;
        } else if (mode == MODE_KROW) {      // [b][h][l][d]
          int b = row >> 8, l = row & 255;
          int h = col >> 6, d = col & 63;
          ((bf16_t*)Cout)[(((size_t)(b * NH + h) * SEQ + l) << 6) + d] = (bf16_t)vv;
        } else {                             // MODE_VT: [b][h][d][l]
          int b = row >> 8, l = row & 255;
          int h = col >> 6, d = col & 63;
          ((bf16_t*)Cout)[(((size_t)(b * NH + h) * DHD + d) << 8) + l] = (bf16_t)vv;
        }
      }
    }
  }
}

// ---------------- attention: per (b,h), S = Q@K^T, softmax, @V ----------------
__global__ __launch_bounds__(128)
void attn_kernel(const bf16_t* __restrict__ Qg,   // [TOK][DIM]
                 const bf16_t* __restrict__ Kgb,  // [b][h][l=256][d=64]
                 const bf16_t* __restrict__ VgT,  // [b][h][d=64][l=256]
                 bf16_t* __restrict__ Att) {      // [TOK][DIM]
  __shared__ alignas(16) float  sS[4][16][256];
  __shared__ alignas(16) bf16_t sP[4][16][256];
  const int pair = blockIdx.x;            // b*NH + h
  const int b = pair >> 3, h = pair & 7;
  const int lane  = threadIdx.x & 31;
  const int wave  = threadIdx.x >> 5;     // 0..3
  const int mrow  = lane & 15;
  const int khalf = lane >> 4;
  const bf16_t* Kp = Kgb + (size_t)pair * SEQ * DHD;
  const bf16_t* Vp = VgT + (size_t)pair * DHD * SEQ;
  const float scale = 0.125f;             // 1/sqrt(64)

  for (int mt = 0; mt < 4; ++mt) {
    const int q0 = (wave * 4 + mt) * 16;
    const bf16_t* Qrow = Qg + (size_t)(b * SEQ + q0 + mrow) * DIM + h * DHD;

    // S panel: 16 x 256
    for (int nt = 0; nt < 16; ++nt) {
      v8f acc = {};
#pragma unroll
      for (int k0 = 0; k0 < DHD; k0 += 32) {
        v16bf av = pack_frag(*(const v8bf*)(Qrow + k0 + (khalf << 3)),
                             *(const v8bf*)(Qrow + k0 + 16 + (khalf << 3)));
        const bf16_t* br = Kp + (size_t)(nt * 16 + mrow) * DHD + k0 + (khalf << 3);
        v16bf bv = pack_frag(*(const v8bf*)br, *(const v8bf*)(br + 16));
        acc = __builtin_amdgcn_wmma_f32_16x16x32_bf16(false, av, false, bv, (short)0, acc, false, false);
      }
#pragma unroll
      for (int r = 0; r < 8; ++r)
        sS[wave][r + (khalf << 3)][nt * 16 + mrow] = acc[r];
    }

    // row softmax (lanes 0..15 each own one query row)
    if (khalf == 0) {
      float mx = -1e30f;
      for (int c = 0; c < 256; ++c) {
        float s = sS[wave][mrow][c] * scale;
        sS[wave][mrow][c] = s;
        mx = fmaxf(mx, s);
      }
      float sum = 0.f;
      for (int c = 0; c < 256; ++c) {
        float e = __expf(sS[wave][mrow][c] - mx);
        sS[wave][mrow][c] = e;
        sum += e;
      }
      float inv = 1.f / sum;
      for (int c = 0; c < 256; ++c)
        sP[wave][mrow][c] = (bf16_t)(sS[wave][mrow][c] * inv);
    }

    // O = P(16x256) @ V(256x64)
    for (int nt = 0; nt < 4; ++nt) {
      v8f acc = {};
      for (int k0 = 0; k0 < SEQ; k0 += 32) {
        const bf16_t* pr = &sP[wave][mrow][k0];
        v16bf av = pack_frag(*(const v8bf*)(pr + (khalf << 3)),
                             *(const v8bf*)(pr + 16 + (khalf << 3)));
        const bf16_t* br = Vp + (size_t)(nt * 16 + mrow) * SEQ + k0 + (khalf << 3);
        v16bf bv = pack_frag(*(const v8bf*)br, *(const v8bf*)(br + 16));
        acc = __builtin_amdgcn_wmma_f32_16x16x32_bf16(false, av, false, bv, (short)0, acc, false, false);
      }
#pragma unroll
      for (int r = 0; r < 8; ++r)
        Att[(size_t)(b * SEQ + q0 + r + (khalf << 3)) * DIM + h * DHD + nt * 16 + mrow] = (bf16_t)acc[r];
    }
  }
}

// ---------------- LayerNorm over last dim 512, y = LN(Xa + Xr) ----------------
__global__ void ln_kernel(const float* __restrict__ Xa, const float* __restrict__ Xr,
                          const float* __restrict__ g, const float* __restrict__ bta,
                          float* __restrict__ Of, bf16_t* __restrict__ Ob, int rows) {
  const int warp = threadIdx.x >> 5;
  const int lane = threadIdx.x & 31;
  const int row = blockIdx.x * (blockDim.x >> 5) + warp;
  if (row >= rows) return;
  const float* xa = Xa + (size_t)row * DIM;
  const float* xr = Xr + (size_t)row * DIM;
  float t[16];
  float s = 0.f;
#pragma unroll
  for (int i = 0; i < 16; ++i) { int d = lane + 32 * i; t[i] = xa[d] + xr[d]; s += t[i]; }
#pragma unroll
  for (int off = 16; off >= 1; off >>= 1) s += __shfl_xor(s, off, 32);
  const float mu = s * (1.f / 512.f);
  float v = 0.f;
#pragma unroll
  for (int i = 0; i < 16; ++i) { float d0 = t[i] - mu; v += d0 * d0; }
#pragma unroll
  for (int off = 16; off >= 1; off >>= 1) v += __shfl_xor(v, off, 32);
  const float rstd = rsqrtf(v * (1.f / 512.f) + 1e-5f);
#pragma unroll
  for (int i = 0; i < 16; ++i) {
    int d = lane + 32 * i;
    float y = (t[i] - mu) * rstd * g[d] + bta[d];
    if (Of) Of[(size_t)row * DIM + d] = y;
    if (Ob) Ob[(size_t)row * DIM + d] = (bf16_t)y;
  }
}

// hn = h (post glb_fc) of last token per batch
__global__ void hn_copy(const float* __restrict__ Hf, float* __restrict__ out) {
  int i = blockIdx.x * blockDim.x + threadIdx.x;
  if (i >= BSZ * DIM) return;
  int b = i >> 9, d = i & 511;
  out[(size_t)BSZ * SEQ * DIM + i] = Hf[((size_t)(b * SEQ + SEQ - 1)) * DIM + d];
}

// ---------------- host-side launch ----------------
static inline int gemm_blocks(int M, int N) { return (M >> 7) * (N >> 6); }

extern "C" void kernel_launch(void* const* d_in, const int* in_sizes, int n_in,
                              void* d_out, int out_size, void* d_ws, size_t ws_size,
                              hipStream_t stream) {
  (void)in_sizes; (void)n_in; (void)out_size; (void)ws_size;
  const float* inp    = (const float*)d_in[0];
  const float* conv1w = (const float*)d_in[1];
  const float* conv2w = (const float*)d_in[2];
  const float* ctx_wq = (const float*)d_in[3];
  // d_in[4] = ctx_wk: mathematically dead (softmax over a single element == 1)
  const float* ctx_fc = (const float*)d_in[5];
  const float* glb_wq = (const float*)d_in[6];
  const float* glb_wk = (const float*)d_in[7];
  const float* glb_fc = (const float*)d_in[8];
  const float* ln1g = (const float*)d_in[9],  *ln1b = (const float*)d_in[10];
  const float* ln2g = (const float*)d_in[11], *ln2b = (const float*)d_in[12];
  const float* ln3g = (const float*)d_in[13], *ln3b = (const float*)d_in[14];
  const float* mlp_w1 = (const float*)d_in[15];
  const float* mlp_b1 = (const float*)d_in[16];
  const float* mlp_w2 = (const float*)d_in[17];
  const float* mlp_b2 = (const float*)d_in[18];
  float* out = (float*)d_out;

  char* ws = (char*)d_ws;
  size_t off = 0;
  auto alloc = [&](size_t bytes) -> void* {
    void* p = ws + off;
    off += (bytes + 255) & ~(size_t)255;
    return p;
  };
  bf16_t* WCTX = (bf16_t*)alloc((size_t)DIM * DIM * 2);          // fused ctx_wq@ctx_fc
  bf16_t* GWQ  = (bf16_t*)alloc((size_t)DIM * DIM * 2);
  bf16_t* GFC  = (bf16_t*)alloc((size_t)DIM * DIM * 2);
  bf16_t* W1B  = (bf16_t*)alloc((size_t)DIM * DFF * 2);
  bf16_t* W2B  = (bf16_t*)alloc((size_t)DFF * DIM * 2);
  bf16_t* KGB  = (bf16_t*)alloc((size_t)BSZ * NH * SEQ * DHD * 2);   // [b][h][l][d]
  bf16_t* VGT  = (bf16_t*)alloc((size_t)BSZ * NH * DHD * SEQ * 2);   // [b][h][d][l]
  bf16_t* INPB = (bf16_t*)alloc((size_t)TOK * DIM * 2);
  float*  BUFA = (float*)alloc((size_t)TOK * DIM * 4);           // CTX, then Hf
  float*  X1F  = (float*)alloc((size_t)TOK * DIM * 4);
  bf16_t* BUFB = (bf16_t*)alloc((size_t)TOK * DIM * 2);          // XCb -> QGb
  bf16_t* BUFC = (bf16_t*)alloc((size_t)TOK * DIM * 2);          // X2b -> ATTb
  float*  X3F  = (float*)alloc((size_t)TOK * DIM * 4);
  bf16_t* X3B  = (bf16_t*)alloc((size_t)TOK * DIM * 2);
  bf16_t* HIDB = (bf16_t*)alloc((size_t)TOK * DFF * 2);
  float*  YF   = (float*)alloc((size_t)TOK * DIM * 4);
  // small early-phase temps overlaid on HIDB region (HIDB written much later)
  bf16_t* T0 = HIDB;
  bf16_t* T1 = HIDB + (size_t)DIM * DIM;
  bf16_t* T2 = HIDB + (size_t)2 * DIM * DIM;

  const int CT = 256;
  const int WW = DIM * DIM;  // 262144

  // --- weight / input conversions ---
  convert_f32_bf16<<<WW / CT, CT, 0, stream>>>(ctx_wq, T0, WW);
  convert_f32_bf16<<<WW / CT, CT, 0, stream>>>(ctx_fc, T1, WW);
  convert_f32_bf16<<<WW / CT, CT, 0, stream>>>(glb_wk, T2, WW);
  convert_f32_bf16<<<WW / CT, CT, 0, stream>>>(glb_wq, GWQ, WW);
  convert_f32_bf16<<<WW / CT, CT, 0, stream>>>(glb_fc, GFC, WW);
  convert_f32_bf16<<<(DIM * DFF) / CT, CT, 0, stream>>>(mlp_w1, W1B, DIM * DFF);
  convert_f32_bf16<<<(DFF * DIM) / CT, CT, 0, stream>>>(mlp_w2, W2B, DFF * DIM);
  convert_f32_bf16<<<(TOK * DIM) / CT, CT, 0, stream>>>(inp, INPB, TOK * DIM);

  // --- fused context weight, global K/V (scatter epilogues) ---
  gemm_wmma<<<gemm_blocks(DIM, DIM), CT, 0, stream>>>(T0, T1, nullptr, WCTX, DIM, DIM, DIM, MODE_BF16);
  gemm_wmma<<<gemm_blocks(TOK, DIM), CT, 0, stream>>>(INPB, T2, nullptr, KGB, TOK, DIM, DIM, MODE_KROW);
  gemm_wmma<<<gemm_blocks(TOK, DIM), CT, 0, stream>>>(INPB, GWQ, nullptr, VGT, TOK, DIM, DIM, MODE_VT);

  // --- token pipeline (all tokens in parallel) ---
  conv3_feature<<<(TOK * DIM) / CT, CT, 0, stream>>>(inp, conv1w, BUFB, TOK);                    // xc
  gemm_wmma<<<gemm_blocks(TOK, DIM), CT, 0, stream>>>(BUFB, WCTX, nullptr, BUFA, TOK, DIM, DIM, MODE_F32); // ctx
  ln_kernel<<<TOK / 8, CT, 0, stream>>>(BUFA, inp, ln1g, ln1b, X1F, nullptr, TOK);               // x1
  conv3_feature<<<(TOK * DIM) / CT, CT, 0, stream>>>(X1F, conv2w, BUFC, TOK);                    // x2
  gemm_wmma<<<gemm_blocks(TOK, DIM), CT, 0, stream>>>(BUFC, GWQ, nullptr, BUFB, TOK, DIM, DIM, MODE_BF16); // Qg
  attn_kernel<<<BSZ * NH, 128, 0, stream>>>(BUFB, KGB, VGT, BUFC);                               // att
  gemm_wmma<<<gemm_blocks(TOK, DIM), CT, 0, stream>>>(BUFC, GFC, nullptr, BUFA, TOK, DIM, DIM, MODE_F32);  // h
  ln_kernel<<<TOK / 8, CT, 0, stream>>>(BUFA, X1F, ln2g, ln2b, X3F, X3B, TOK);                   // x3
  gemm_wmma<<<gemm_blocks(TOK, DFF), CT, 0, stream>>>(X3B, W1B, mlp_b1, HIDB, TOK, DFF, DIM, MODE_GELU);
  gemm_wmma<<<gemm_blocks(TOK, DIM), CT, 0, stream>>>(HIDB, W2B, mlp_b2, YF, TOK, DIM, DFF, MODE_F32BIAS);
  ln_kernel<<<TOK / 8, CT, 0, stream>>>(YF, X3F, ln3g, ln3b, out, TOK > 0 ? nullptr : nullptr, TOK); // output
  hn_copy<<<(BSZ * DIM) / CT, CT, 0, stream>>>(BUFA, out);                                       // hn tail
}
